// NSE_16458314678389
// MI455X (gfx1250) — compile-verified, matrix operand order checked by reading, not run on hardware
//
#include <hip/hip_runtime.h>
#include <hip/hip_bf16.h>

// ---------------------------------------------------------------------------
// NSE forward for MI455X (gfx1250): wave32, WMMA bf16 16x16x32 GEMMs,
// per-example memory M kept bf16 (39MB -> L2-resident), batch split 16 rows
// per workgroup (WMMA M-dim), 512-step recurrence per WG.
// ---------------------------------------------------------------------------

typedef __attribute__((ext_vector_type(16))) __bf16 v16bf;
typedef __attribute__((ext_vector_type(8)))  float  v8f;
typedef __attribute__((ext_vector_type(8)))  unsigned int v8u;

#define Bsz 128
#define Lsz 512
#define Nsz 300
#define N4  1200
#define MLPsz 1024

// K layouts (padded to multiples of 32):
//   read GEMM : [x(0..299) | pad | rh(320..619) | pad]        K = 640
//   comp GEMM : [m_rt(0..299) | pad | rh(320..619) | pad]     K = 640
//   write GEMM: [c_t(0..599) | pad | wh(640..939) | pad]      K = 960
#define KR 640
#define KW 960
#define NTR 76   /* 1216/16 */
#define KTR 20
#define NTC 38   /* 608/16 */
#define KTC 20
#define NTW 76
#define KTW 30
#define GSTR 1216  /* gate buffer row stride */

__device__ __forceinline__ unsigned short f2bf(float f) {
    union { float f; unsigned int u; } x; x.f = f;
    unsigned int u = x.u;
    unsigned int r = u + 0x7fffu + ((u >> 16) & 1u);
    if ((u & 0x7f800000u) == 0x7f800000u) r = u;   // inf/nan: truncate
    return (unsigned short)(r >> 16);
}
__device__ __forceinline__ float bf2f(unsigned short s) {
    union { unsigned int u; float f; } x; x.u = ((unsigned int)s) << 16;
    return x.f;
}
__device__ __forceinline__ float sigm(float x) { return 1.0f / (1.0f + __expf(-x)); }

// ---------------------------------------------------------------------------
// WMMA GEMM: D(16 x NT*16) = A(16 x KT*32, bf16 in LDS) x B(tiled bf16 global)
// A-frag : lane (l&15)=row, (l>>4) selects K-half; 16 contiguous bf16 per lane.
// B-frag : pre-packed tiles of 32 lanes x 16 bf16 (32B contiguous per lane).
// D      : lane (l&15)=col, rows (l>>4)*8 + j.
// ---------------------------------------------------------------------------
template<int KT, int NT, int ASTR, int DSTR, int NCAP>
__device__ __forceinline__ void wg_gemm_bf16(const unsigned short* __restrict__ A,
                                             const unsigned short* __restrict__ Bg,
                                             const float* __restrict__ bias,
                                             unsigned short* __restrict__ D,
                                             int wid, int lane) {
    const int row   = lane & 15;
    const int khalf = lane >> 4;
    const int col   = lane & 15;
    for (int nt = wid; nt < NT; nt += 16) {
        v8f acc = {};
#pragma unroll
        for (int kt = 0; kt < KT; ++kt) {
            const v16bf a = *(const v16bf*)(A + row * ASTR + kt * 32 + khalf * 16);
            const unsigned short* bp = Bg + ((size_t)(kt * NT + nt) * 32 + lane) * 16;
            if (kt + 1 < KT)
                __builtin_prefetch(Bg + ((size_t)((kt + 1) * NT + nt) * 32 + lane) * 16, 0, 0);
            const v16bf b = *(const v16bf*)bp;
            acc = __builtin_amdgcn_wmma_f32_16x16x32_bf16(false, a, false, b,
                                                          (short)0, acc, false, false);
        }
        const int n  = nt * 16 + col;
        const float bv = bias[n];
#pragma unroll
        for (int j = 0; j < 8; ++j) {
            const int m = khalf * 8 + j;
            if (n < NCAP) D[m * DSTR + n] = f2bf(acc[j] + bv);
        }
    }
}

// ---------------------------------------------------------------------------
// Weight packing into WMMA B-fragment tile layout (bf16).
// B[k][n] assembled from up to two fp32 row-major sources W[n][c] (B = W^T).
// ---------------------------------------------------------------------------
__global__ void nse_pack_b(unsigned short* __restrict__ dst,
                           const float* __restrict__ s0, int ld0, int co0, int k0, int len0,
                           const float* __restrict__ s1, int ld1, int co1, int k1, int len1,
                           int Nsrc, int KT, int NT) {
    int gid = blockIdx.x * blockDim.x + threadIdx.x;
    int total = KT * NT * 512;
    if (gid >= total) return;
    int j    = gid & 15;
    int lane = (gid >> 4) & 31;
    int tile = gid >> 9;
    int nt = tile % NT;
    int kt = tile / NT;
    int k = kt * 32 + (lane >> 4) * 16 + j;
    int n = nt * 16 + (lane & 15);
    float v = 0.0f;
    if (n < Nsrc) {
        if (k >= k0 && k < k0 + len0)       v = s0[(size_t)n * ld0 + co0 + (k - k0)];
        else if (k >= k1 && k < k1 + len1)  v = s1[(size_t)n * ld1 + co1 + (k - k1)];
    }
    dst[gid] = f2bf(v);
}

__global__ void nse_pack_bias(float* __restrict__ dst, const float* __restrict__ a,
                              const float* __restrict__ b, int n, int ncap) {
    int i = blockIdx.x * blockDim.x + threadIdx.x;
    if (i >= ncap) return;
    float v = 0.0f;
    if (i < n) { v = a[i]; if (b) v += b[i]; }
    dst[i] = v;
}

// M0[b][n][l] = embed[inputs[b][l]][n]  (bf16)
__global__ void nse_init_M(unsigned short* __restrict__ M,
                           const int* __restrict__ inputs,
                           const float* __restrict__ embed) {
    int gid = blockIdx.x * blockDim.x + threadIdx.x;
    if (gid >= Bsz * Nsz * Lsz) return;
    int l   = gid & (Lsz - 1);
    int bn  = gid >> 9;
    int n   = bn % Nsz;
    int b   = bn / Nsz;
    int tok = inputs[b * Lsz + l];
    M[gid] = f2bf(embed[(size_t)tok * Nsz + n]);
}

// ---------------------------------------------------------------------------
// Main recurrence: 8 WGs x 512 threads (16 waves). WG w owns batch rows
// [16w, 16w+16). One wave per batch row for attention / softmax / M update.
// ---------------------------------------------------------------------------
__global__ __launch_bounds__(512, 1)
void nse_recurrence(const int* __restrict__ inputs,
                    const int* __restrict__ seqlen,
                    const float* __restrict__ embed,
                    const unsigned short* __restrict__ Br,
                    const unsigned short* __restrict__ Bc,
                    const unsigned short* __restrict__ Bw,
                    const float* __restrict__ rbias,
                    const float* __restrict__ cbias,
                    const float* __restrict__ wbias,
                    unsigned short* __restrict__ M,
                    float* __restrict__ outh) {
    extern __shared__ unsigned char smem[];
    unsigned short* xa = (unsigned short*)smem;      // 16 x 640 : [x|rh] then [m_rt|rh]
    unsigned short* ca = xa + 16 * KR;               // 16 x 960 : [c_t|pad|wh]
    unsigned short* gb = ca + 16 * KW;               // 16 x 1216: read/write-LSTM gates

    const int tid  = threadIdx.x;
    const int wid  = tid >> 5;
    const int lane = tid & 31;
    const int b0   = blockIdx.x * 16;

    for (int i = tid; i < 16 * KR; i += 512) xa[i] = 0;
    for (int i = tid; i < 16 * KW; i += 512) ca[i] = 0;

    float rc[10], wc[10];
#pragma unroll
    for (int s = 0; s < 10; ++s) { rc[s] = 0.0f; wc[s] = 0.0f; }
    float zr[16];   // per-lane softmax weights for this wave's batch row

    unsigned short* Mrow = M + (size_t)(b0 + wid) * Nsz * Lsz;
    __syncthreads();

    for (int t = 0; t < Lsz; ++t) {
        // ---- 1. gather x_t into xa[r][0..299] (bf16) -----------------------
        for (int idx = tid; idx < 16 * Nsz; idx += 512) {
            int r = idx / Nsz, n = idx - r * Nsz;
            int tok = inputs[(b0 + r) * Lsz + t];
            xa[r * KR + n] = f2bf(embed[(size_t)tok * Nsz + n]);
        }
        __syncthreads();

        // ---- 2. read-LSTM gates: [x|rh] x Br -> gb -------------------------
        wg_gemm_bf16<KTR, NTR, KR, GSTR, GSTR>(xa, Br, rbias, gb, wid, lane);
        __syncthreads();

        // ---- 3. read-LSTM activations: rh -> xa[r][320+n], rc in regs ------
#pragma unroll
        for (int s = 0; s < 10; ++s) {
            int idx = s * 512 + tid;
            if (idx < 16 * Nsz) {
                int r = idx / Nsz, n = idx - r * Nsz;
                float gi = bf2f(gb[r * GSTR + n]);
                float gf = bf2f(gb[r * GSTR + 300 + n]);
                float gg = bf2f(gb[r * GSTR + 600 + n]);
                float go = bf2f(gb[r * GSTR + 900 + n]);
                float c  = sigm(gf) * rc[s] + sigm(gi) * tanhf(gg);
                rc[s] = c;
                xa[r * KR + 320 + n] = f2bf(sigm(go) * tanhf(c));
            }
        }
        __syncthreads();

        // ---- 4. attention: one wave per row; z in registers ----------------
        {
            const int r = wid;
#pragma unroll
            for (int c = 0; c < 16; ++c) zr[c] = 0.0f;
            for (int n = 0; n < Nsz; ++n) {
                float hn = bf2f(xa[r * KR + 320 + n]);
                v8u mv = *(const v8u*)(Mrow + (size_t)n * Lsz + lane * 16);
#pragma unroll
                for (int cc = 0; cc < 8; ++cc) {
                    unsigned int u = mv[cc];
                    zr[2 * cc]     += hn * bf2f((unsigned short)(u & 0xffffu));
                    zr[2 * cc + 1] += hn * bf2f((unsigned short)(u >> 16));
                }
            }
            float mx = -3.402823e38f;
#pragma unroll
            for (int c = 0; c < 16; ++c) mx = fmaxf(mx, zr[c]);
#pragma unroll
            for (int off = 16; off >= 1; off >>= 1) mx = fmaxf(mx, __shfl_xor(mx, off, 32));
            float sum = 0.0f;
#pragma unroll
            for (int c = 0; c < 16; ++c) { zr[c] = __expf(zr[c] - mx); sum += zr[c]; }
#pragma unroll
            for (int off = 16; off >= 1; off >>= 1) sum += __shfl_xor(sum, off, 32);
            float inv = 1.0f / sum;
#pragma unroll
            for (int c = 0; c < 16; ++c) zr[c] *= inv;

            // m_rt[n] = sum_l z[l] * M[n][l] -> xa[r][0..299] (reuses x region)
            for (int n = 0; n < Nsz; ++n) {
                v8u mv = *(const v8u*)(Mrow + (size_t)n * Lsz + lane * 16);
                float p = 0.0f;
#pragma unroll
                for (int cc = 0; cc < 8; ++cc) {
                    unsigned int u = mv[cc];
                    p += zr[2 * cc]     * bf2f((unsigned short)(u & 0xffffu));
                    p += zr[2 * cc + 1] * bf2f((unsigned short)(u >> 16));
                }
#pragma unroll
                for (int off = 16; off >= 1; off >>= 1) p += __shfl_xor(p, off, 32);
                if (lane == 0) xa[r * KR + n] = f2bf(p);
            }
        }
        __syncthreads();

        // ---- 5. compose: [m_rt|rh] x Bc -> c_t in ca[r][0..599] ------------
        wg_gemm_bf16<KTC, NTC, KR, KW, 600>(xa, Bc, cbias, ca, wid, lane);
        __syncthreads();

        // ---- 6. write-LSTM gates: [c_t|wh] x Bw -> gb ----------------------
        wg_gemm_bf16<KTW, NTW, KW, GSTR, GSTR>(ca, Bw, wbias, gb, wid, lane);
        __syncthreads();

        // ---- 7. write-LSTM activations: wh -> ca[r][640+n]; grab out_h -----
#pragma unroll
        for (int s = 0; s < 10; ++s) {
            int idx = s * 512 + tid;
            if (idx < 16 * Nsz) {
                int r = idx / Nsz, n = idx - r * Nsz;
                float gi = bf2f(gb[r * GSTR + n]);
                float gf = bf2f(gb[r * GSTR + 300 + n]);
                float gg = bf2f(gb[r * GSTR + 600 + n]);
                float go = bf2f(gb[r * GSTR + 900 + n]);
                float c  = sigm(gf) * wc[s] + sigm(gi) * tanhf(gg);
                wc[s] = c;
                float h = sigm(go) * tanhf(c);
                ca[r * KW + 640 + n] = f2bf(h);
                if (t == seqlen[b0 + r] - 1) outh[(b0 + r) * Nsz + n] = h;
            }
        }
        __syncthreads();

        // ---- 8. memory update: M = (1-z)*M + wh*z --------------------------
        {
            const int r = wid;
            for (int n = 0; n < Nsz; ++n) {
                float whn = bf2f(ca[r * KW + 640 + n]);
                unsigned short* mp = Mrow + (size_t)n * Lsz + lane * 16;
                v8u mv = *(const v8u*)mp;
                v8u ov;
#pragma unroll
                for (int cc = 0; cc < 8; ++cc) {
                    unsigned int u = mv[cc];
                    float a = bf2f((unsigned short)(u & 0xffffu));
                    float b = bf2f((unsigned short)(u >> 16));
                    float z0 = zr[2 * cc], z1 = zr[2 * cc + 1];
                    unsigned int lo = f2bf((1.0f - z0) * a + whn * z0);
                    unsigned int hi = f2bf((1.0f - z1) * b + whn * z1);
                    ov[cc] = lo | (hi << 16);
                }
                *(v8u*)mp = ov;
            }
        }
        __syncthreads();
    }
}

// ---------------------------------------------------------------------------
// Head: out = relu(out_h @ fc1^T + b1) @ fc2^T + b2   (128 x 2)
// ---------------------------------------------------------------------------
__global__ void nse_head(const float* __restrict__ outh,
                         const float* __restrict__ fc1W, const float* __restrict__ fc1b,
                         const float* __restrict__ fc2W, const float* __restrict__ fc2b,
                         float* __restrict__ out) {
    __shared__ float h[Nsz];
    __shared__ float red0[256], red1[256];
    const int b = blockIdx.x, tid = threadIdx.x;
    for (int i = tid; i < Nsz; i += 256) h[i] = outh[b * Nsz + i];
    __syncthreads();
    float p0 = 0.0f, p1 = 0.0f;
    for (int j = tid; j < MLPsz; j += 256) {
        float a = fc1b[j];
        const float* wrow = fc1W + (size_t)j * Nsz;
        for (int n = 0; n < Nsz; ++n) a += h[n] * wrow[n];
        a = fmaxf(a, 0.0f);
        p0 += a * fc2W[j];
        p1 += a * fc2W[MLPsz + j];
    }
    red0[tid] = p0; red1[tid] = p1;
    __syncthreads();
    for (int s = 128; s >= 1; s >>= 1) {
        if (tid < s) { red0[tid] += red0[tid + s]; red1[tid] += red1[tid + s]; }
        __syncthreads();
    }
    if (tid == 0) {
        out[b * 2 + 0] = red0[0] + fc2b[0];
        out[b * 2 + 1] = red1[0] + fc2b[1];
    }
}

// ---------------------------------------------------------------------------
extern "C" void kernel_launch(void* const* d_in, const int* in_sizes, int n_in,
                              void* d_out, int out_size, void* d_ws, size_t ws_size,
                              hipStream_t stream) {
    const int*   inputs  = (const int*)d_in[0];
    const int*   seqlen  = (const int*)d_in[1];
    const float* embed   = (const float*)d_in[2];
    const float* rW_ih   = (const float*)d_in[3];
    const float* rW_hh   = (const float*)d_in[4];
    const float* rb_ih   = (const float*)d_in[5];
    const float* rb_hh   = (const float*)d_in[6];
    const float* wW_ih   = (const float*)d_in[7];
    const float* wW_hh   = (const float*)d_in[8];
    const float* wb_ih   = (const float*)d_in[9];
    const float* wb_hh   = (const float*)d_in[10];
    const float* comp_W  = (const float*)d_in[11];
    const float* comp_b  = (const float*)d_in[12];
    const float* fc1_W   = (const float*)d_in[13];
    const float* fc1_b   = (const float*)d_in[14];
    const float* fc2_W   = (const float*)d_in[15];
    const float* fc2_b   = (const float*)d_in[16];

    unsigned char* ws = (unsigned char*)d_ws;
    size_t off = 0;
    auto carve = [&](size_t bytes) { size_t r = off; off += (bytes + 255) & ~(size_t)255; return r; };
    unsigned short* Br    = (unsigned short*)(ws + carve((size_t)KR * NTR * 16 * 2));
    unsigned short* Bc    = (unsigned short*)(ws + carve((size_t)KR * NTC * 16 * 2));
    unsigned short* Bw    = (unsigned short*)(ws + carve((size_t)KW * NTW * 16 * 2));
    float*          rbias = (float*)(ws + carve(NTR * 16 * 4));
    float*          cbias = (float*)(ws + carve(NTC * 16 * 4));
    float*          wbias = (float*)(ws + carve(NTW * 16 * 4));
    unsigned short* Mmem  = (unsigned short*)(ws + carve((size_t)Bsz * Nsz * Lsz * 2));
    float*          outh  = (float*)(ws + carve((size_t)Bsz * Nsz * 4));

    // ---- pack weights into WMMA B-fragment tiles (bf16) --------------------
    {
        int th = 256;
        int n1 = KTR * NTR * 512;  // read: [rW_ih @ k0..299 | rW_hh @ k320..619], N=1200
        nse_pack_b<<<(n1 + th - 1) / th, th, 0, stream>>>(Br,
            rW_ih, Nsz, 0, 0, Nsz, rW_hh, Nsz, 0, 320, Nsz, N4, KTR, NTR);
        int n2 = KTC * NTC * 512;  // comp: m_rt part (cols 300..599) @ k0, rh part @ k320
        nse_pack_b<<<(n2 + th - 1) / th, th, 0, stream>>>(Bc,
            comp_W, 2 * Nsz, Nsz, 0, Nsz, comp_W, 2 * Nsz, 0, 320, Nsz, 2 * Nsz, KTC, NTC);
        int n3 = KTW * NTW * 512;  // write: [wW_ih @ k0..599 | wW_hh @ k640..939], N=1200
        nse_pack_b<<<(n3 + th - 1) / th, th, 0, stream>>>(Bw,
            wW_ih, 2 * Nsz, 0, 0, 2 * Nsz, wW_hh, Nsz, 0, 640, Nsz, N4, KTW, NTW);
        nse_pack_bias<<<(NTR * 16 + th - 1) / th, th, 0, stream>>>(rbias, rb_ih, rb_hh, N4, NTR * 16);
        nse_pack_bias<<<(NTC * 16 + th - 1) / th, th, 0, stream>>>(cbias, comp_b, nullptr, 2 * Nsz, NTC * 16);
        nse_pack_bias<<<(NTW * 16 + th - 1) / th, th, 0, stream>>>(wbias, wb_ih, wb_hh, N4, NTW * 16);
        int nm = Bsz * Nsz * Lsz;
        nse_init_M<<<(nm + th - 1) / th, th, 0, stream>>>(Mmem, inputs, embed);
    }

    // ---- 512-step recurrence: 8 WGs x 512 threads, ~88KB dynamic LDS -------
    size_t lds = (size_t)(16 * KR + 16 * KW + 16 * GSTR) * 2;
    nse_recurrence<<<Bsz / 16, 512, lds, stream>>>(inputs, seqlen, embed,
                                                   Br, Bc, Bw, rbias, cbias, wbias,
                                                   Mmem, outh);

    // ---- MLP head ----------------------------------------------------------
    nse_head<<<Bsz, 256, 0, stream>>>(outh, fc1_W, fc1_b, fc2_W, fc2_b, (float*)d_out);
}
